// TinyRNNLM_33097017983594
// MI455X (gfx1250) — compile-verified, matrix operand order checked by reading, not run on hardware
//
#include <hip/hip_runtime.h>
#include <hip/hip_bf16.h>

// ---------------------------------------------------------------------------
// TinyRNNLM on MI455X (gfx1250): GRU + LM head with v_wmma_f32_16x16x32_f16.
// Roofline: 412MB fp32 logits write -> ~18us floor @ 23.3TB/s; LM head GEMM
// (52.7 GFLOP) runs as f16 WMMA out of L2 (hs 1MB + W_out-f16 25.7MB << 192MB).
// Logits are streamed with non-temporal stores so the 412MB write-once stream
// does not evict the reused GEMM operands from L2.
// GRU recurrence: M = batch = 16 == WMMA M; one WGP, h in LDS, W_hh-f16 in L2.
// ---------------------------------------------------------------------------

#define V_VOCAB   50257
#define V_PAD     50304   // 393 * 128
#define D_MODEL   128
#define H_HID     256
#define G3        768     // 3*H
#define BATCH     16
#define SEQ       128
#define BT        2048    // BATCH*SEQ

typedef _Float16 v16h __attribute__((ext_vector_type(16)));
typedef _Float16 v8h  __attribute__((ext_vector_type(8)));
typedef float    v8f  __attribute__((ext_vector_type(8)));

// Workspace layout (bytes, 256-aligned)
#define OFF_GX     0u                                   // BT*768 f32   = 6,291,456
#define OFF_WOUT16 6291456u                             // V_PAD*256 f16 = 25,755,648
#define OFF_WHH16  (OFF_WOUT16 + 25755648u)             // 768*256 f16  = 393,216
#define OFF_HS16   (OFF_WHH16 + 393216u)                // BT*256 f16   = 1,048,576

// ---------------------------------------------------------------------------
// f16 fragment loader for row-major-[rows][ld] data with K contiguous.
// ISA 7.12.2 (16-bit A 16x32): lanes 0-15: M=lane, halves 0-7 <-> K=kb..kb+7,
// halves 8-15 <-> K=kb+16..kb+23; lanes 16-31: same with kb += 8.
// Works for A (rows=M) and B (rows=N, since B[k][n] = W[n][k] row-major in k).
// ---------------------------------------------------------------------------
__device__ __forceinline__ v16h load_frag(const _Float16* __restrict__ base,
                                          int row, int ld, int kb) {
  const _Float16* p = base + row * ld + kb;
  v8h lo = *(const v8h*)(p);
  v8h hi = *(const v8h*)(p + 16);
  return __builtin_shufflevector(lo, hi,
      0, 1, 2, 3, 4, 5, 6, 7, 8, 9, 10, 11, 12, 13, 14, 15);
}

// ---------------------------------------------------------------------------
// Kernel 1: f32 -> f16 convert with zero-padding (W_out padded to V_PAD rows,
// so head-GEMM B-fragment loads never run off the buffer). Source is read
// exactly once -> non-temporal load; f16 destination is reused -> RT store.
// ---------------------------------------------------------------------------
__global__ void cvt_f16_kernel(const float* __restrict__ src,
                               _Float16* __restrict__ dst,
                               int n_src, int n_total) {
  int i = blockIdx.x * blockDim.x + threadIdx.x;
  int stride = gridDim.x * blockDim.x;
  for (; i < n_total; i += stride) {
    float v = (i < n_src) ? __builtin_nontemporal_load(src + i) : 0.0f;
    dst[i] = (_Float16)v;
  }
}

// ---------------------------------------------------------------------------
// Kernel 2: embedding gather + input-side gate projection.
// gx[t][b][g] = sum_d embed[ids[b,t]][d] * W_ih[g][d] + b_ih[g]
// 256 blocks x 8 (b,t) rows each; x rows staged in LDS, W_ih streamed from L2.
// ---------------------------------------------------------------------------
__global__ __launch_bounds__(256) void embed_proj_kernel(
    const int* __restrict__ ids, const float* __restrict__ embed,
    const float* __restrict__ Wih, const float* __restrict__ bih,
    float* __restrict__ gx) {
  __shared__ float xs[8][D_MODEL];
  const int tid = threadIdx.x;
  const int p = blockIdx.x;  // 0..255, handles pairs p*8 .. p*8+7 (b-major)
#pragma unroll
  for (int i = 0; i < 4; ++i) {
    int idx = tid + 256 * i;             // 0..1023
    int r = idx >> 7, d = idx & 127;
    int id = ids[p * 8 + r];
    xs[r][d] = embed[(size_t)id * D_MODEL + d];
  }
  __syncthreads();
#pragma unroll
  for (int gs = 0; gs < 3; ++gs) {
    int g = tid + gs * 256;              // gate column 0..767
    float bias = bih[g];
    float acc[8];
#pragma unroll
    for (int r = 0; r < 8; ++r) acc[r] = bias;
    const float* w = Wih + g * D_MODEL;
    for (int d = 0; d < D_MODEL; ++d) {
      float wv = w[d];
#pragma unroll
      for (int r = 0; r < 8; ++r) acc[r] += wv * xs[r][d];
    }
#pragma unroll
    for (int r = 0; r < 8; ++r) {
      int pair = p * 8 + r;
      int b = pair >> 7, t = pair & 127;
      gx[(t * BATCH + b) * G3 + g] = acc[r];  // [t][b][3H] for GRU step access
    }
  }
}

// ---------------------------------------------------------------------------
// Kernel 3: GRU recurrence. Single workgroup (32 waves) on one WGP.
// Per step: gh = h @ W_hh^T + b_hh via WMMA (M=16=batch, N=768, K=256):
// 24 waves x 2 N-tiles x 8 k-chunks of v_wmma_f32_16x16x32_f16.
// Then all 32 waves do the gate nonlinearities elementwise.
// ---------------------------------------------------------------------------
__global__ __launch_bounds__(1024) void gru_kernel(
    const float* __restrict__ gx, const _Float16* __restrict__ whh16,
    const float* __restrict__ bhh, _Float16* __restrict__ hs16) {
  __shared__ float    hF[BATCH * H_HID];      // 16 KB
  __shared__ _Float16 h16[BATCH * H_HID];     // 8 KB (WMMA A operand)
  __shared__ float    gh[BATCH * G3];         // 48 KB

  const int tid = threadIdx.x;
  for (int i = tid; i < BATCH * H_HID; i += 1024) {
    hF[i] = 0.0f;
    h16[i] = (_Float16)0.0f;
  }
  __syncthreads();

  const int wave = tid >> 5;
  const int lane = tid & 31;
  const int rrow = lane & 15;                 // M (batch) / N row within tile
  const int kb_off = (lane & 16) >> 1;        // 0 or 8
  const int mbase = (lane & 16) ? 8 : 0;      // C/D row base for this half-wave

  for (int t = 0; t < SEQ; ++t) {
    if (wave < 24) {                          // uniform per wave: EXEC all-1s
#pragma unroll
      for (int s = 0; s < 2; ++s) {
        const int nt = wave * 2 + s;          // N-tile 0..47
        const int col = nt * 16 + rrow;       // gate column
        v8f acc = {};
#pragma unroll
        for (int kc = 0; kc < 8; ++kc) {
          const int kb = kc * 32 + kb_off;
          v16h a = load_frag(h16, rrow, H_HID, kb);     // LDS (ds_load_b128)
          v16h b = load_frag(whh16, col, H_HID, kb);    // L2  (global_load_b128)
          acc = __builtin_amdgcn_wmma_f32_16x16x32_f16(
              false, a, false, b, (short)0, acc, false, false);
        }
        const float bias = bhh[col];
#pragma unroll
        for (int j = 0; j < 8; ++j)
          gh[(mbase + j) * G3 + col] = acc[j] + bias;
      }
    }
    __syncthreads();

    const float* gxt = gx + (size_t)t * (BATCH * G3);
#pragma unroll
    for (int i = 0; i < 4; ++i) {
      const int e = tid + i * 1024;           // 0..4095 unique
      const int b = e >> 8, j = e & 255;
      const float xr = gxt[b * G3 + j];
      const float xz = gxt[b * G3 + 256 + j];
      const float xn = gxt[b * G3 + 512 + j];
      const float hr = gh[b * G3 + j];
      const float hz = gh[b * G3 + 256 + j];
      const float hn = gh[b * G3 + 512 + j];
      const float hp = hF[b * H_HID + j];
      const float r = 1.0f / (1.0f + __expf(-(xr + hr)));
      const float z = 1.0f / (1.0f + __expf(-(xz + hz)));
      const float u = xn + r * hn;
      const float e2 = __expf(-2.0f * fabsf(u));
      const float n = copysignf((1.0f - e2) / (1.0f + e2), u);
      const float hnew = (1.0f - z) * n + z * hp;
      hF[b * H_HID + j] = hnew;
      h16[b * H_HID + j] = (_Float16)hnew;
      hs16[((size_t)(b * SEQ + t)) * H_HID + j] = (_Float16)hnew;  // [B*T][H]
    }
    __syncthreads();
  }
}

// ---------------------------------------------------------------------------
// Kernel 4: LM head GEMM. out[m][v] = sum_k hs16[m][k] * wout16[v][k].
// Block = 128(M) x 128(N) tile; 8 waves in a 4x2 grid, each wave 32x64
// (2x4 WMMA tiles, 8 k-chunks -> 64 wmma per wave).
// Operands stay L2-resident; the 412MB write-once logits stream bypasses
// cache retention via non-temporal stores (TH=NT) so it cannot evict them.
// ---------------------------------------------------------------------------
__global__ __launch_bounds__(256) void head_kernel(
    const _Float16* __restrict__ hs16, const _Float16* __restrict__ wout16,
    float* __restrict__ out) {
  const int lane = threadIdx.x & 31;
  const int wave = threadIdx.x >> 5;
  const int Moff = blockIdx.y * 128 + (wave >> 1) * 32;
  const int Noff = blockIdx.x * 128 + (wave & 1) * 64;
  const int rrow = lane & 15;
  const int kb_off = (lane & 16) >> 1;
  const int mbase = (lane & 16) ? 8 : 0;

  v8f acc[2][4];
  v8f zero = {};
#pragma unroll
  for (int mt = 0; mt < 2; ++mt)
#pragma unroll
    for (int nt = 0; nt < 4; ++nt) acc[mt][nt] = zero;

  for (int kc = 0; kc < 8; ++kc) {
    const int kb = kc * 32 + kb_off;
    // hint the next k-chunk of the B stream into cache
    if (kc < 7)
      __builtin_prefetch(wout16 + (size_t)(Noff + rrow) * H_HID + kb + 32, 0, 0);
    v16h a0 = load_frag(hs16, Moff + rrow, H_HID, kb);
    v16h a1 = load_frag(hs16, Moff + 16 + rrow, H_HID, kb);
    v16h b0 = load_frag(wout16, Noff + 0 + rrow, H_HID, kb);
    v16h b1 = load_frag(wout16, Noff + 16 + rrow, H_HID, kb);
    v16h b2 = load_frag(wout16, Noff + 32 + rrow, H_HID, kb);
    v16h b3 = load_frag(wout16, Noff + 48 + rrow, H_HID, kb);
    acc[0][0] = __builtin_amdgcn_wmma_f32_16x16x32_f16(false, a0, false, b0, (short)0, acc[0][0], false, false);
    acc[0][1] = __builtin_amdgcn_wmma_f32_16x16x32_f16(false, a0, false, b1, (short)0, acc[0][1], false, false);
    acc[0][2] = __builtin_amdgcn_wmma_f32_16x16x32_f16(false, a0, false, b2, (short)0, acc[0][2], false, false);
    acc[0][3] = __builtin_amdgcn_wmma_f32_16x16x32_f16(false, a0, false, b3, (short)0, acc[0][3], false, false);
    acc[1][0] = __builtin_amdgcn_wmma_f32_16x16x32_f16(false, a1, false, b0, (short)0, acc[1][0], false, false);
    acc[1][1] = __builtin_amdgcn_wmma_f32_16x16x32_f16(false, a1, false, b1, (short)0, acc[1][1], false, false);
    acc[1][2] = __builtin_amdgcn_wmma_f32_16x16x32_f16(false, a1, false, b2, (short)0, acc[1][2], false, false);
    acc[1][3] = __builtin_amdgcn_wmma_f32_16x16x32_f16(false, a1, false, b3, (short)0, acc[1][3], false, false);
  }

#pragma unroll
  for (int mt = 0; mt < 2; ++mt) {
#pragma unroll
    for (int nt = 0; nt < 4; ++nt) {
      const int col = Noff + nt * 16 + rrow;
      if (col < V_VOCAB) {
        const int mrow = Moff + mt * 16 + mbase;
#pragma unroll
        for (int j = 0; j < 8; ++j)
          __builtin_nontemporal_store(
              acc[mt][nt][j], out + (size_t)(mrow + j) * V_VOCAB + col);
      }
    }
  }
}

// ---------------------------------------------------------------------------
extern "C" void kernel_launch(void* const* d_in, const int* in_sizes, int n_in,
                              void* d_out, int out_size, void* d_ws,
                              size_t ws_size, hipStream_t stream) {
  (void)in_sizes; (void)n_in; (void)out_size; (void)ws_size;
  const int*   ids   = (const int*)d_in[0];
  const float* embed = (const float*)d_in[1];
  const float* Wih   = (const float*)d_in[2];
  const float* bih   = (const float*)d_in[3];
  const float* Whh   = (const float*)d_in[4];
  const float* bhh   = (const float*)d_in[5];
  const float* Wout  = (const float*)d_in[6];
  float* out = (float*)d_out;

  char* ws = (char*)d_ws;
  float*    gx     = (float*)(ws + OFF_GX);
  _Float16* wout16 = (_Float16*)(ws + OFF_WOUT16);
  _Float16* whh16  = (_Float16*)(ws + OFF_WHH16);
  _Float16* hs16   = (_Float16*)(ws + OFF_HS16);

  // 1) precision conversion (W_out zero-padded to V_PAD rows)
  cvt_f16_kernel<<<1024, 256, 0, stream>>>(Wout, wout16, V_VOCAB * H_HID,
                                           V_PAD * H_HID);
  cvt_f16_kernel<<<96, 256, 0, stream>>>(Whh, whh16, G3 * H_HID, G3 * H_HID);
  // 2) embedding + input projection
  embed_proj_kernel<<<256, 256, 0, stream>>>(ids, embed, Wih, bih, gx);
  // 3) sequential GRU (single workgroup, WMMA with M = batch = 16)
  gru_kernel<<<1, 1024, 0, stream>>>(gx, whh16, bhh, hs16);
  // 4) LM head: 393 x 16 blocks of 128x128 output tiles
  head_kernel<<<dim3(393, 16), 256, 0, stream>>>(hs16, wout16, out);
}
